// MinkUNet_57019985821719
// MI455X (gfx1250) — matrix operand order repaired and checked by reading, not compiled, not run on hardware
//
#include <hip/hip_runtime.h>

// ---------------------------------------------------------------------------
// MinkUNet-style dense 3D U-Net on a 48^3 grid, for MI455X (gfx1250).
// Convs = implicit GEMM on V_WMMA_F32_16X16X32_F16 (wave32).
// Each wave computes a 32-voxel x 32-Cout register tile (4 accumulators),
// so every (tap, cin-chunk) step issues 8 b128 loads feeding 4 WMMAs.
// ---------------------------------------------------------------------------

#define VSIZE 0.05f
#define GRIDD 48

typedef _Float16 half_t;
typedef _Float16 v16h __attribute__((ext_vector_type(16)));
typedef _Float16 v8h  __attribute__((ext_vector_type(8)));
typedef float    v8f  __attribute__((ext_vector_type(8)));

// ------------------------------ utility kernels ----------------------------

__global__ void k_voxel_scatter(const float* __restrict__ feats, const float* __restrict__ coords,
                                float* __restrict__ vsum, float* __restrict__ vcnt, int N) {
  int i = blockIdx.x * blockDim.x + threadIdx.x;
  if (i >= N) return;
  const float inv = 1.0f / VSIZE;
  int ix = (int)floorf(coords[i * 3 + 0] * inv);
  int iy = (int)floorf(coords[i * 3 + 1] * inv);
  int iz = (int)floorf(coords[i * 3 + 2] * inv);
  ix = min(max(ix, 0), GRIDD - 1);
  iy = min(max(iy, 0), GRIDD - 1);
  iz = min(max(iz, 0), GRIDD - 1);
  int flat = (ix * GRIDD + iy) * GRIDD + iz;
  atomicAdd(&vcnt[flat], 1.0f);
#pragma unroll
  for (int c = 0; c < 4; ++c)
    atomicAdd(&vsum[(long)flat * 4 + c], feats[(long)i * 4 + c]);
}

// write scatter-mean result into channel-padded (stride 32) f16 buffer
__global__ void k_voxel_finalize(const float* __restrict__ vsum, const float* __restrict__ vcnt,
                                 half_t* __restrict__ x0, long nvox) {
  long i = (long)blockIdx.x * blockDim.x + threadIdx.x;
  if (i >= nvox * 32) return;
  const int c = (int)(i & 31);
  const long v = i >> 5;
  float val = 0.f;
  if (c < 4) val = vsum[v * 4 + c] / fmaxf(vcnt[v], 1.0f);
  x0[i] = (half_t)val;
}

// -------------------- weight pre-pack into B-fragment order ----------------
// Fragment f = (tap*nChunk + cc)*nCtPad + ct; within a fragment, element
// (lane, e) = W[tap'][cinBase + 8*(lane>>4) + (e&7) + 16*(e>>3)][ct*16 + (lane&15)],
// zero-padded past Cin/Cout, spatially flipped when flip!=0 (transposed conv).

__global__ void k_pack_w(const float* __restrict__ W, half_t* __restrict__ out,
                         int KV, int Cin, int Cout, int nChunk, int nCtPad, int flip) {
  const long total = (long)KV * nChunk * nCtPad * 512;
  const long i = (long)blockIdx.x * blockDim.x + threadIdx.x;
  if (i >= total) return;
  const int e = (int)(i & 15);
  const int lane = (int)((i >> 4) & 31);
  const long f = i >> 9;
  const int ct = (int)(f % nCtPad);
  const int cc = (int)((f / nCtPad) % nChunk);
  const int t = (int)(f / ((long)nCtPad * nChunk));
  const int tw = flip ? (KV - 1 - t) : t;
  const int ci = (cc << 5) + ((lane >> 4) << 3) + (e & 7) + ((e >> 3) << 4);
  const int co = ct * 16 + (lane & 15);
  half_t v = (half_t)0.f;
  if (ci < Cin && co < Cout) v = (half_t)W[((long)tw * Cin + ci) * Cout + co];
  out[i] = v;
}

// ---------------------- WMMA implicit-GEMM 3D convolution ------------------
// Block = 128 threads = 4 independent waves. Wave tile: 32 output voxels
// (two 16-row M tiles) x 32 output channels (two 16-wide N tiles).
// A: clamped-address b128 loads + branchless zeroing for out-of-grid taps.
// B: b128 loads from the pre-packed fragment stream (single pointer bump).

__global__ __launch_bounds__(128) void k_conv3d_wmma(
    const half_t* __restrict__ X,   // [Din^3, Cp] f16, Cp = channel stride (mult of 32)
    const half_t* __restrict__ Wp,  // packed fragments
    float* __restrict__ Y,          // [Dout^3, Cout] f32 raw conv output
    int Din, int Cp, int Dout, int Cout, int K, int stride, int pad, int nCtPad) {
  const int lane = threadIdx.x & 31;
  const int wave = threadIdx.x >> 5;
  const int hid = lane >> 4;
  const int ln = lane & 15;
  const long Vout = (long)Dout * Dout * Dout;
  const long voxTile = ((long)blockIdx.x * 4 + wave) * 32;
  const int nChunk = Cp >> 5;

  // two A-row voxels per lane (M tile 0 and M tile 1)
  const long vx0 = voxTile + ln;
  const long vx1 = voxTile + 16 + ln;
  const int ok0 = (vx0 < Vout), ok1 = (vx1 < Vout);
  const long m0 = ok0 ? vx0 : 0, m1 = ok1 ? vx1 : 0;
  const int ow0 = (int)(m0 % Dout), oh0 = (int)((m0 / Dout) % Dout),
            od0 = (int)(m0 / ((long)Dout * Dout));
  const int ow1 = (int)(m1 % Dout), oh1 = (int)((m1 / Dout) % Dout),
            od1 = (int)(m1 / ((long)Dout * Dout));

  v8f acc00 = {0.f, 0.f, 0.f, 0.f, 0.f, 0.f, 0.f, 0.f};
  v8f acc01 = acc00, acc10 = acc00, acc11 = acc00;
  const v8h zero8 = {};
  const int KV = K * K * K;

  // per-lane B fragment pointer for the first N tile; second tile = +512
  const half_t* wfrag = Wp + ((long)blockIdx.y * 2 * 512) + (long)lane * 16;
  const long fragStride = (long)nCtPad * 512;

  for (int t = 0; t < KV; ++t) {
    const int kd = t / (K * K), kh = (t / K) % K, kw = t % K;
    // tile-0 input tap
    const int id0 = od0 * stride + kd - pad;
    const int ih0 = oh0 * stride + kh - pad;
    const int iw0 = ow0 * stride + kw - pad;
    const int in0 = ok0 & ((unsigned)id0 < (unsigned)Din) &
                    ((unsigned)ih0 < (unsigned)Din) & ((unsigned)iw0 < (unsigned)Din);
    const half_t* xp0 =
        X + (((long)min(max(id0, 0), Din - 1) * Din + min(max(ih0, 0), Din - 1)) * Din +
             min(max(iw0, 0), Din - 1)) * Cp + hid * 8;
    // tile-1 input tap
    const int id1 = od1 * stride + kd - pad;
    const int ih1 = oh1 * stride + kh - pad;
    const int iw1 = ow1 * stride + kw - pad;
    const int in1 = ok1 & ((unsigned)id1 < (unsigned)Din) &
                    ((unsigned)ih1 < (unsigned)Din) & ((unsigned)iw1 < (unsigned)Din);
    const half_t* xp1 =
        X + (((long)min(max(id1, 0), Din - 1) * Din + min(max(ih1, 0), Din - 1)) * Din +
             min(max(iw1, 0), Din - 1)) * Cp + hid * 8;

    for (int cc = 0; cc < nChunk; ++cc) {
      v8h a00 = *(const v8h*)(xp0 + (cc << 5));
      v8h a01 = *(const v8h*)(xp0 + (cc << 5) + 16);
      if (!in0) { a00 = zero8; a01 = zero8; }
      const v16h a0 = __builtin_shufflevector(a00, a01, 0, 1, 2, 3, 4, 5, 6, 7,
                                              8, 9, 10, 11, 12, 13, 14, 15);
      v8h a10 = *(const v8h*)(xp1 + (cc << 5));
      v8h a11 = *(const v8h*)(xp1 + (cc << 5) + 16);
      if (!in1) { a10 = zero8; a11 = zero8; }
      const v16h a1 = __builtin_shufflevector(a10, a11, 0, 1, 2, 3, 4, 5, 6, 7,
                                              8, 9, 10, 11, 12, 13, 14, 15);

      const v8h b00 = *(const v8h*)wfrag;
      const v8h b01 = *(const v8h*)(wfrag + 8);
      const v8h b10 = *(const v8h*)(wfrag + 512);
      const v8h b11 = *(const v8h*)(wfrag + 520);
      wfrag += fragStride;
      const v16h b0 = __builtin_shufflevector(b00, b01, 0, 1, 2, 3, 4, 5, 6, 7,
                                              8, 9, 10, 11, 12, 13, 14, 15);
      const v16h b1 = __builtin_shufflevector(b10, b11, 0, 1, 2, 3, 4, 5, 6, 7,
                                              8, 9, 10, 11, 12, 13, 14, 15);

      acc00 = __builtin_amdgcn_wmma_f32_16x16x32_f16(false, a0, false, b0, (short)0, acc00, false, false);
      acc01 = __builtin_amdgcn_wmma_f32_16x16x32_f16(false, a0, false, b1, (short)0, acc01, false, false);
      acc10 = __builtin_amdgcn_wmma_f32_16x16x32_f16(false, a1, false, b0, (short)0, acc10, false, false);
      acc11 = __builtin_amdgcn_wmma_f32_16x16x32_f16(false, a1, false, b1, (short)0, acc11, false, false);
    }
  }

  // D layout: lane col N=ln, VGPR r holds row M = r + 8*hid
  const int co0 = blockIdx.y * 32 + ln;
  const int co1 = co0 + 16;
  const int w0 = (co0 < Cout), w1 = (co1 < Cout);
#pragma unroll
  for (int r = 0; r < 8; ++r) {
    const long vA = voxTile + (r + 8 * hid);
    const long vB = vA + 16;
    if (vA < Vout) {
      if (w0) Y[vA * Cout + co0] = acc00[r];
      if (w1) Y[vA * Cout + co1] = acc01[r];
    }
    if (vB < Vout) {
      if (w0) Y[vB * Cout + co0] = acc10[r];
      if (w1) Y[vB * Cout + co1] = acc11[r];
    }
  }
}

// ------------------------------ batch norm ---------------------------------

__global__ void k_bn_stats(const float* __restrict__ Y, float* __restrict__ sums,
                           long n, int C) {
  __shared__ float ssum[192], ssq[192];
  for (int c = threadIdx.x; c < C; c += blockDim.x) { ssum[c] = 0.f; ssq[c] = 0.f; }
  __syncthreads();
  for (long i = (long)blockIdx.x * blockDim.x + threadIdx.x; i < n;
       i += (long)gridDim.x * blockDim.x) {
    const float v = Y[i];
    const int c = (int)(i % C);
    atomicAdd(&ssum[c], v);
    atomicAdd(&ssq[c], v * v);
  }
  __syncthreads();
  for (int c = threadIdx.x; c < C; c += blockDim.x) {
    atomicAdd(&sums[c], ssum[c]);
    atomicAdd(&sums[C + c], ssq[c]);
  }
}

// normalize + affine (+residual) (+ReLU); writes unpadded f32 and padded f16
__global__ void k_bn_apply(const float* __restrict__ raw, const float* __restrict__ stats,
                           const float* __restrict__ g, const float* __restrict__ b,
                           const float* __restrict__ resid,
                           float* __restrict__ o32, half_t* __restrict__ o16,
                           long nvox, int C, int Cp, int relu) {
  const long n = nvox * Cp;
  const long i = (long)blockIdx.x * blockDim.x + threadIdx.x;
  if (i >= n) return;
  const int c = (int)(i % Cp);
  const long v = i / Cp;
  if (c >= C) {               // zero the pad lanes of the f16 buffer
    if (o16) o16[i] = (half_t)0.f;
    return;
  }
  const long j = v * C + c;
  const float invn = 1.0f / (float)nvox;
  const float m = stats[c] * invn;
  const float var = stats[C + c] * invn - m * m;
  float y = (raw[j] - m) * rsqrtf(var + 1e-5f) * g[c] + b[c];
  if (resid) y += resid[j];
  if (relu) y = fmaxf(y, 0.f);
  if (o32) o32[j] = y;
  if (o16) o16[i] = (half_t)y;
}

// --------------------------- concat / dilation -----------------------------

__global__ void k_concat2(const float* __restrict__ A, int Ca,
                          const float* __restrict__ B, int Cb,
                          half_t* __restrict__ out, long nvox) {
  const int C = Ca + Cb;   // always a multiple of 32 in this net
  const long n = nvox * C;
  const long i = (long)blockIdx.x * blockDim.x + threadIdx.x;
  if (i >= n) return;
  const long v = i / C;
  const int c = (int)(i % C);
  const float x = (c < Ca) ? A[v * Ca + c] : B[v * Cb + (c - Ca)];
  out[i] = (half_t)x;
}

// zero-dilate by 2x for transposed conv (out buffer must be pre-zeroed)
__global__ void k_dilate2x(const float* __restrict__ in, half_t* __restrict__ out,
                           int Din, int C) {
  const long n = (long)Din * Din * Din * C;
  const long i = (long)blockIdx.x * blockDim.x + threadIdx.x;
  if (i >= n) return;
  const int c = (int)(i % C);
  const long v = i / C;
  const int iw = (int)(v % Din);
  const int ih = (int)((v / Din) % Din);
  const int id = (int)(v / ((long)Din * Din));
  const int Do = Din * 2;
  const long ov = (((long)id * 2 * Do) + ih * 2) * Do + iw * 2;
  out[ov * C + c] = (half_t)in[i];
}

// ------------------ trilinear gather + L2 normalize ------------------------
// One wave per point; lane == channel (32 channels <-> wave32 exactly).

__global__ void k_gather_norm(const float* __restrict__ grid, const float* __restrict__ bias,
                              const float* __restrict__ coords, float* __restrict__ out,
                              int N) {
  const int lane = threadIdx.x & 31;
  const int wid = blockIdx.x * (blockDim.x >> 5) + (threadIdx.x >> 5);
  if (wid >= N) return;
  const float inv = 1.0f / VSIZE;
  const float fx = coords[(long)wid * 3 + 0] * inv;
  const float fy = coords[(long)wid * 3 + 1] * inv;
  const float fz = coords[(long)wid * 3 + 2] * inv;
  const float bx = floorf(fx), by = floorf(fy), bz = floorf(fz);
  const float rx = fx - bx, ry = fy - by, rz = fz - bz;
  const int ix = min(max((int)bx, 0), GRIDD - 1);
  const int iy = min(max((int)by, 0), GRIDD - 1);
  const int iz = min(max((int)bz, 0), GRIDD - 1);
  float acc = 0.f;
#pragma unroll
  for (int d = 0; d < 8; ++d) {
    const int dx = (d >> 2) & 1, dy = (d >> 1) & 1, dz = d & 1;
    const int cx = min(ix + dx, GRIDD - 1);
    const int cy = min(iy + dy, GRIDD - 1);
    const int cz = min(iz + dz, GRIDD - 1);
    const float w = (dx ? rx : 1.f - rx) * (dy ? ry : 1.f - ry) * (dz ? rz : 1.f - rz);
    acc += grid[(((long)cx * GRIDD + cy) * GRIDD + cz) * 32 + lane] * w;
  }
  acc += bias[lane];   // trilinear weights sum to 1 -> bias added post-gather
  float ss = acc * acc;
#pragma unroll
  for (int m = 16; m > 0; m >>= 1) ss += __shfl_xor(ss, m, 32);
  out[(long)wid * 32 + lane] = acc * rsqrtf(ss);
}

// ------------------------------ host driver --------------------------------

namespace {

struct Arena {
  char* p;
  size_t off;
  void* get(size_t bytes) {
    size_t a = (off + 255) & ~(size_t)255;
    off = a + bytes;
    return p + a;
  }
};

inline unsigned cdiv(long a, long b) { return (unsigned)((a + b - 1) / b); }
inline int nct_pad(int Cout) { return ((Cout / 16) + 1) & ~1; }  // even # of 16-tiles

void conv_l(hipStream_t s, const half_t* x, const half_t* wp, float* y,
            int Din, int Cp, int Dout, int Cout, int K, int stride, int pad) {
  const long Vout = (long)Dout * Dout * Dout;
  const int nCt = nct_pad(Cout);
  dim3 g(cdiv(Vout, 128), (unsigned)(nCt / 2), 1);
  k_conv3d_wmma<<<g, dim3(128), 0, s>>>(x, wp, y, Din, Cp, Dout, Cout, K, stride, pad, nCt);
}

void bn_l(hipStream_t s, float* stats, const float* raw,
          const float* g, const float* b, const float* resid,
          float* o32, half_t* o16, long nvox, int C, int Cp, int relu) {
  hipMemsetAsync(stats, 0, 2 * (size_t)C * sizeof(float), s);
  const long ns = nvox * C;
  unsigned sb = cdiv(ns, 256);
  if (sb > 2048) sb = 2048;
  k_bn_stats<<<sb, 256, 0, s>>>(raw, stats, ns, C);
  const long na = nvox * (o16 ? Cp : C);
  k_bn_apply<<<cdiv(na, 256), 256, 0, s>>>(raw, stats, g, b, resid, o32, o16,
                                           nvox, C, o16 ? Cp : C, relu);
}

}  // namespace

extern "C" void kernel_launch(void* const* d_in, const int* in_sizes, int n_in,
                              void* d_out, int out_size, void* d_ws, size_t ws_size,
                              hipStream_t stream) {
  (void)n_in; (void)out_size; (void)ws_size;
  const float* feats  = (const float*)d_in[0];   // [N,4]
  const float* coords = (const float*)d_in[1];   // [N,3]
  const int N = in_sizes[0] / 4;
  auto P = [&](int i) { return (const float*)d_in[i]; };

  // params flattened depth-first in dict insertion order; each conv = w,g,b
  // 5..7 d1.conv  8..10 d1.c1  11..13 d1.c2
  // 14..16 d2.conv 17..19 d2.c1 20..22 d2.c2
  // 23..25 d3.conv 26..28 d3.c1 29..31 d3.c2
  // 32..34 d4.conv 35..37 d4.c1 38..40 d4.c2
  // 41..43 bo.deconv 44..46 bo.c1 47..49 bo.c2
  // 50..52 u1.c1 53..55 u1.c2 56..58 u1.sc 59..61 u1.deconv
  // 62..64 u2.c1 65..67 u2.c2 68..70 u2.sc 71..73 u2.deconv
  // 74..76 u3.conv  77 final.w  78 final.b

  Arena A{(char*)d_ws, 0};
  const long V48 = 48L * 48 * 48, V24 = 24L * 24 * 24, V12 = 12L * 12 * 12, V6 = 6L * 6 * 6;

  auto packw = [&](int idx, int KV, int Cin, int Cout, int flip) -> half_t* {
    const int nCh = (Cin + 31) >> 5;
    const int nCt = nct_pad(Cout);
    const long n = (long)KV * nCh * nCt * 512;
    half_t* p = (half_t*)A.get((size_t)n * sizeof(half_t));
    k_pack_w<<<cdiv(n, 256), 256, 0, stream>>>(P(idx), p, KV, Cin, Cout, nCh, nCt, flip);
    return p;
  };
  auto fb = [&](long n) { return (float*)A.get((size_t)n * sizeof(float)); };
  auto hb = [&](long n) { return (half_t*)A.get((size_t)n * sizeof(half_t)); };

  // packed f16 weights (flip baked in for transposed convs)
  half_t* w_d1 = packw(5, 125, 4, 16, 0);
  half_t* w_d1c1 = packw(8, 27, 16, 16, 0), * w_d1c2 = packw(11, 27, 16, 16, 0);
  half_t* w_d2 = packw(14, 27, 16, 32, 0);
  half_t* w_d2c1 = packw(17, 27, 32, 32, 0), * w_d2c2 = packw(20, 27, 32, 32, 0);
  half_t* w_d3 = packw(23, 27, 32, 64, 0);
  half_t* w_d3c1 = packw(26, 27, 64, 64, 0), * w_d3c2 = packw(29, 27, 64, 64, 0);
  half_t* w_d4 = packw(32, 27, 64, 128, 0);
  half_t* w_d4c1 = packw(35, 27, 128, 128, 0), * w_d4c2 = packw(38, 27, 128, 128, 0);
  half_t* w_bod = packw(41, 27, 128, 128, 1);
  half_t* w_boc1 = packw(44, 27, 128, 128, 0), * w_boc2 = packw(47, 27, 128, 128, 0);
  half_t* w_u1c1 = packw(50, 27, 192, 128, 0), * w_u1c2 = packw(53, 27, 128, 128, 0);
  half_t* w_u1sc = packw(56, 1, 192, 128, 0);
  half_t* w_u1d = packw(59, 27, 128, 64, 1);
  half_t* w_u2c1 = packw(62, 27, 96, 64, 0), * w_u2c2 = packw(65, 27, 64, 64, 0);
  half_t* w_u2sc = packw(68, 1, 96, 64, 0);
  half_t* w_u2d = packw(71, 27, 64, 48, 1);
  half_t* w_u3 = packw(74, 27, 64, 48, 0);
  half_t* w_fin = packw(77, 1, 48, 32, 0);

  // activations / scratch (f16 buffers use channel stride padded to mult-of-32)
  float* stats = fb(2 * 192);
  float* vsum = fb(V48 * 4);
  float* vcnt = fb(V48);
  half_t* x0 = hb(V48 * 32);                 // Cp=32 (4 real)
  float* raw = fb(V48 * 48);                 // reused raw conv output (max 48^3 x 48)

  float* s1a32 = fb(V48 * 16); half_t* s1a16 = hb(V48 * 32); half_t* t48 = hb(V48 * 32);
  float* s1_32 = fb(V48 * 16); half_t* s1_16 = hb(V48 * 32);
  float* s2a32 = fb(V24 * 32); half_t* s2a16 = hb(V24 * 32); half_t* t24 = hb(V24 * 32);
  float* s2_32 = fb(V24 * 32); half_t* s2_16 = hb(V24 * 32);
  float* s4a32 = fb(V12 * 64); half_t* s4a16 = hb(V12 * 64); half_t* t12a = hb(V12 * 64);
  float* s4_32 = fb(V12 * 64); half_t* s4_16 = hb(V12 * 64);
  float* s8a32 = fb(V6 * 128); half_t* s8a16 = hb(V6 * 128); half_t* t6 = hb(V6 * 128);
  float* s8_32 = fb(V6 * 128);
  half_t* dil12 = hb(V12 * 128);
  float* b4a32 = fb(V12 * 128); half_t* b4a16 = hb(V12 * 128); half_t* t12b = hb(V12 * 128);
  float* b4_32 = fb(V12 * 128);
  half_t* cat12 = hb(V12 * 192); half_t* u1t = hb(V12 * 128);
  float* sc12 = fb(V12 * 128); float* r1_32 = fb(V12 * 128);
  half_t* dil24 = hb(V24 * 128);
  float* u1_32 = fb(V24 * 64);
  half_t* cat24 = hb(V24 * 96); half_t* u2t = hb(V24 * 64);
  float* sc24 = fb(V24 * 64); float* r2_32 = fb(V24 * 64);
  half_t* dil48 = hb(V48 * 64);
  float* u2_32 = fb(V48 * 48);
  half_t* cat48 = hb(V48 * 64);
  half_t* u3_16 = hb(V48 * 64);              // Cp=64 (48 real)

  // ---- voxelize (scatter-mean) ----
  hipMemsetAsync(vsum, 0, (size_t)V48 * 4 * sizeof(float), stream);
  hipMemsetAsync(vcnt, 0, (size_t)V48 * sizeof(float), stream);
  k_voxel_scatter<<<cdiv(N, 256), 256, 0, stream>>>(feats, coords, vsum, vcnt, N);
  k_voxel_finalize<<<cdiv(V48 * 32, 256), 256, 0, stream>>>(vsum, vcnt, x0, V48);

  // ---- down1 (48^3, 4->16) ----
  conv_l(stream, x0, w_d1, raw, 48, 32, 48, 16, 5, 1, 2);
  bn_l(stream, stats, raw, P(6), P(7), nullptr, s1a32, s1a16, V48, 16, 32, 1);
  conv_l(stream, s1a16, w_d1c1, raw, 48, 32, 48, 16, 3, 1, 1);
  bn_l(stream, stats, raw, P(9), P(10), nullptr, nullptr, t48, V48, 16, 32, 1);
  conv_l(stream, t48, w_d1c2, raw, 48, 32, 48, 16, 3, 1, 1);
  bn_l(stream, stats, raw, P(12), P(13), s1a32, s1_32, s1_16, V48, 16, 32, 1);

  // ---- down2 (48->24, 16->32) ----
  conv_l(stream, s1_16, w_d2, raw, 48, 32, 24, 32, 3, 2, 1);
  bn_l(stream, stats, raw, P(15), P(16), nullptr, s2a32, s2a16, V24, 32, 32, 1);
  conv_l(stream, s2a16, w_d2c1, raw, 24, 32, 24, 32, 3, 1, 1);
  bn_l(stream, stats, raw, P(18), P(19), nullptr, nullptr, t24, V24, 32, 32, 1);
  conv_l(stream, t24, w_d2c2, raw, 24, 32, 24, 32, 3, 1, 1);
  bn_l(stream, stats, raw, P(21), P(22), s2a32, s2_32, s2_16, V24, 32, 32, 1);

  // ---- down3 (24->12, 32->64) ----
  conv_l(stream, s2_16, w_d3, raw, 24, 32, 12, 64, 3, 2, 1);
  bn_l(stream, stats, raw, P(24), P(25), nullptr, s4a32, s4a16, V12, 64, 64, 1);
  conv_l(stream, s4a16, w_d3c1, raw, 12, 64, 12, 64, 3, 1, 1);
  bn_l(stream, stats, raw, P(27), P(28), nullptr, nullptr, t12a, V12, 64, 64, 1);
  conv_l(stream, t12a, w_d3c2, raw, 12, 64, 12, 64, 3, 1, 1);
  bn_l(stream, stats, raw, P(30), P(31), s4a32, s4_32, s4_16, V12, 64, 64, 1);

  // ---- down4 (12->6, 64->128) ----
  conv_l(stream, s4_16, w_d4, raw, 12, 64, 6, 128, 3, 2, 1);
  bn_l(stream, stats, raw, P(33), P(34), nullptr, s8a32, s8a16, V6, 128, 128, 1);
  conv_l(stream, s8a16, w_d4c1, raw, 6, 128, 6, 128, 3, 1, 1);
  bn_l(stream, stats, raw, P(36), P(37), nullptr, nullptr, t6, V6, 128, 128, 1);
  conv_l(stream, t6, w_d4c2, raw, 6, 128, 6, 128, 3, 1, 1);
  bn_l(stream, stats, raw, P(39), P(40), s8a32, s8_32, nullptr, V6, 128, 128, 1);

  // ---- bottleneck: deconv 6->12 (zero-dilate + flipped conv), res(128) ----
  hipMemsetAsync(dil12, 0, (size_t)V12 * 128 * sizeof(half_t), stream);
  k_dilate2x<<<cdiv(V6 * 128, 256), 256, 0, stream>>>(s8_32, dil12, 6, 128);
  conv_l(stream, dil12, w_bod, raw, 12, 128, 12, 128, 3, 1, 1);
  bn_l(stream, stats, raw, P(42), P(43), nullptr, b4a32, b4a16, V12, 128, 128, 1);
  conv_l(stream, b4a16, w_boc1, raw, 12, 128, 12, 128, 3, 1, 1);
  bn_l(stream, stats, raw, P(45), P(46), nullptr, nullptr, t12b, V12, 128, 128, 1);
  conv_l(stream, t12b, w_boc2, raw, 12, 128, 12, 128, 3, 1, 1);
  bn_l(stream, stats, raw, P(48), P(49), b4a32, b4_32, nullptr, V12, 128, 128, 1);

  // ---- up1: cat(b4,s4)=192 -> res(192->128) -> deconv 12->24 (128->64) ----
  k_concat2<<<cdiv(V12 * 192, 256), 256, 0, stream>>>(b4_32, 128, s4_32, 64, cat12, V12);
  conv_l(stream, cat12, w_u1sc, raw, 12, 192, 12, 128, 1, 1, 0);
  bn_l(stream, stats, raw, P(57), P(58), nullptr, sc12, nullptr, V12, 128, 128, 0);
  conv_l(stream, cat12, w_u1c1, raw, 12, 192, 12, 128, 3, 1, 1);
  bn_l(stream, stats, raw, P(51), P(52), nullptr, nullptr, u1t, V12, 128, 128, 1);
  conv_l(stream, u1t, w_u1c2, raw, 12, 128, 12, 128, 3, 1, 1);
  bn_l(stream, stats, raw, P(54), P(55), sc12, r1_32, nullptr, V12, 128, 128, 1);
  hipMemsetAsync(dil24, 0, (size_t)V24 * 128 * sizeof(half_t), stream);
  k_dilate2x<<<cdiv(V12 * 128, 256), 256, 0, stream>>>(r1_32, dil24, 12, 128);
  conv_l(stream, dil24, w_u1d, raw, 24, 128, 24, 64, 3, 1, 1);
  bn_l(stream, stats, raw, P(60), P(61), nullptr, u1_32, nullptr, V24, 64, 64, 1);

  // ---- up2: cat(u1,s2)=96 -> res(96->64) -> deconv 24->48 (64->48) ----
  k_concat2<<<cdiv(V24 * 96, 256), 256, 0, stream>>>(u1_32, 64, s2_32, 32, cat24, V24);
  conv_l(stream, cat24, w_u2sc, raw, 24, 96, 24, 64, 1, 1, 0);
  bn_l(stream, stats, raw, P(69), P(70), nullptr, sc24, nullptr, V24, 64, 64, 0);
  conv_l(stream, cat24, w_u2c1, raw, 24, 96, 24, 64, 3, 1, 1);
  bn_l(stream, stats, raw, P(63), P(64), nullptr, nullptr, u2t, V24, 64, 64, 1);
  conv_l(stream, u2t, w_u2c2, raw, 24, 64, 24, 64, 3, 1, 1);
  bn_l(stream, stats, raw, P(66), P(67), sc24, r2_32, nullptr, V24, 64, 64, 1);
  hipMemsetAsync(dil48, 0, (size_t)V48 * 64 * sizeof(half_t), stream);
  k_dilate2x<<<cdiv(V24 * 64, 256), 256, 0, stream>>>(r2_32, dil48, 24, 64);
  conv_l(stream, dil48, w_u2d, raw, 48, 64, 48, 48, 3, 1, 1);
  bn_l(stream, stats, raw, P(72), P(73), nullptr, u2_32, nullptr, V48, 48, 48, 1);

  // ---- up3: cat(u2,s1)=64 -> conv 64->48 ----
  k_concat2<<<cdiv(V48 * 64, 256), 256, 0, stream>>>(u2_32, 48, s1_32, 16, cat48, V48);
  conv_l(stream, cat48, w_u3, raw, 48, 64, 48, 48, 3, 1, 1);
  bn_l(stream, stats, raw, P(75), P(76), nullptr, nullptr, u3_16, V48, 48, 64, 1);

  // ---- final 1x1x1 conv 48->32 (raw output; bias folded into gather) ----
  conv_l(stream, u3_16, w_fin, raw, 48, 64, 48, 32, 1, 1, 0);

  // ---- trilinear gather + L2 normalize (wave per point, lane = channel) ----
  k_gather_norm<<<cdiv(N, 8), 256, 0, stream>>>(raw, P(78), coords, (float*)d_out, N);
}
